// LongformerAttentionBlock_28449863368948
// MI455X (gfx1250) — matrix operand order repaired and checked by reading, not compiled, not run on hardware
//
#include <hip/hip_runtime.h>
#include <hip/hip_bf16.h>

#define BB 2
#define CC 96
#define IMGS 256
#define NS 128
#define NN 16384
#define TD 384
#define HH 6
#define DD 64
#define W1 128
#define NC 128
#define GG 50
#define GP 64
#define NEGV (-1.0e9f)

typedef unsigned short u16;
typedef __bf16 bf16_t;
typedef bf16_t v16bf __attribute__((ext_vector_type(16)));
typedef bf16_t v8bf  __attribute__((ext_vector_type(8)));
typedef float  v8f   __attribute__((ext_vector_type(8)));
typedef int    v4i   __attribute__((ext_vector_type(4)));

__device__ __forceinline__ u16 f2bf(float x) {
  unsigned u = __float_as_uint(x);
  u = (u + 0x7FFFu + ((u >> 16) & 1u)) >> 16;
  return (u16)u;
}

__device__ __forceinline__ v8f zero8() {
  v8f z = {0.f,0.f,0.f,0.f,0.f,0.f,0.f,0.f};
  return z;
}

// A-fragment (16x32 bf16): lane m holds K = [hi*8, hi*8+8) and [16+hi*8, 16+hi*8+8)
__device__ __forceinline__ v16bf load_a16(const u16* row, int hi) {
  v8bf lo = *(const v8bf*)(row + hi * 8);
  v8bf hb = *(const v8bf*)(row + 16 + hi * 8);
  v16bf r;
#pragma unroll
  for (int i = 0; i < 8; ++i) { r[i] = lo[i]; r[i + 8] = hb[i]; }
  return r;
}

// B-fragment (32x16 bf16): lane n holds column n, K = [hi*8, hi*8+8) and [16+hi*8, 16+hi*8+8)
__device__ __forceinline__ v16bf load_b16(const u16* colrun, int hi) {
  v8bf lo = *(const v8bf*)(colrun + hi * 8);
  v8bf hb = *(const v8bf*)(colrun + 16 + hi * 8);
  v16bf r;
#pragma unroll
  for (int i = 0; i < 8; ++i) { r[i] = lo[i]; r[i + 8] = hb[i]; }
  return r;
}

__device__ __forceinline__ v8f wmma_bf(v16bf a, v16bf b, v8f c) {
  return __builtin_amdgcn_wmma_f32_16x16x32_bf16(false, a, false, b, (short)0, c, false, false);
}

__device__ __forceinline__ int gpos_of(int g) {
  return (int)((double)g * (double)(NN - 1) / (double)(GG - 1));
}

// ---- async global->LDS copy of 16 bytes (tracked by ASYNCcnt), with fallback ----
__device__ __forceinline__ void async_copy16(const u16* g, u16* l) {
#if defined(__gfx1250__) && __has_builtin(__builtin_amdgcn_global_load_async_to_lds_b128)
  __builtin_amdgcn_global_load_async_to_lds_b128((v4i*)g, (v4i*)l, 0, 0);
#else
  *(v8bf*)l = *(const v8bf*)g;
#endif
}

__device__ __forceinline__ void wait_async0() {
#if defined(__gfx1250__)
#if __has_builtin(__builtin_amdgcn_s_wait_asynccnt)
  __builtin_amdgcn_s_wait_asynccnt(0);
#else
  asm volatile("s_wait_asynccnt 0" ::: "memory");
#endif
#endif
}

// ---------------- prep: transpose weights to bf16 (N-major rows of K) ----------------
__global__ void prep_weights(const float* w0, const float* w1, const float* w2,
                             const float* w3, const float* w4, const float* w5,
                             u16* wt) {
  int idx = blockIdx.x * blockDim.x + threadIdx.x;   // 6*TD*TD
  int mat = idx / (TD * TD);
  int r   = idx % (TD * TD);
  int n = r / TD, k = r % TD;
  const float* src = mat == 0 ? w0 : mat == 1 ? w1 : mat == 2 ? w2
                   : mat == 3 ? w3 : mat == 4 ? w4 : w5;
  wt[(size_t)mat * TD * TD + (size_t)n * TD + k] = f2bf(src[(size_t)k * TD + n]);
}

// ---------------- tokenize x -> bf16 tokens (B,N,TD) ----------------
__global__ void tokenize(const float* x, u16* tok) {
  int idx = blockIdx.x * blockDim.x + threadIdx.x;   // B*N*TD
  int td = idx % TD;
  int rest = idx / TD;
  int n = rest % NN;
  int b = rest / NN;
  int i = n / NS, j = n % NS;
  int c = td >> 2;
  int w = td & 3;
  int wi = w >> 1, wj = w & 1;
  float v = x[(((size_t)b * CC + c) * IMGS + (i * 2 + wi)) * IMGS + (j * 2 + wj)];
  tok[idx] = f2bf(v);
}

// ---------------- 5 full projections: q(scaled), k, vT, kga, vgaT ----------------
__global__ void proj_gemm(const u16* tok, const u16* wt,
                          const float* bq, const float* bk, const float* bv,
                          const float* bkg, const float* bvg,
                          u16* q, u16* k, u16* vT, u16* kga, u16* vgaT) {
  int wave = blockIdx.x * 8 + (threadIdx.x >> 5);
  int lane = threadIdx.x & 31;
  int col = lane & 15, hi = lane >> 4;
  int p   = wave / (2048 * 6);
  int rem = wave % (2048 * 6);
  int mt = rem / 6, ng = rem % 6;
  int m0 = mt * 16, n0 = ng * 64;
  int wsel = (p < 3) ? p : p + 1;     // skip Wqg at slot 3
  const u16* A  = tok + (size_t)(m0 + col) * TD;
  const u16* Wm = wt + (size_t)wsel * TD * TD;

  v8f acc[4];
#pragma unroll
  for (int t = 0; t < 4; ++t) acc[t] = zero8();

  for (int kk = 0; kk < TD; kk += 32) {
    v16bf a = load_a16(A + kk, hi);
#pragma unroll
    for (int t = 0; t < 4; ++t) {
      v16bf b = load_b16(Wm + (size_t)(n0 + t * 16 + col) * TD + kk, hi);
      acc[t] = wmma_bf(a, b, acc[t]);
    }
  }

  const float* bias = p == 0 ? bq : p == 1 ? bk : p == 2 ? bv : p == 3 ? bkg : bvg;
  float scale = (p == 0) ? 0.125f : 1.0f;
#pragma unroll
  for (int t = 0; t < 4; ++t) {
#pragma unroll
    for (int r = 0; r < 8; ++r) {
      int ncol = n0 + t * 16 + col;
      int m = m0 + r + hi * 8;
      int b_ = m >> 14;
      int nn = m & (NN - 1);
      int d = ncol & 63;
      float val = (acc[t][r] + bias[ncol]) * scale;
      u16 bfv = f2bf(val);
      size_t bh = (size_t)b_ * HH + ng;
      if (p == 0)      q[(bh * NN + nn) * DD + d] = bfv;
      else if (p == 1) k[(bh * NN + nn) * DD + d] = bfv;
      else if (p == 2) vT[(bh * DD + d) * NN + nn] = bfv;
      else if (p == 3) kga[(bh * NN + nn) * DD + d] = bfv;
      else             vgaT[(bh * DD + d) * NN + nn] = bfv;
    }
  }
}

// ---------------- qg = tokens[GPOS] @ Wqg, scaled; (B,H,GP,D) bf16 ----------------
__global__ void qg_gemm(const u16* tok, const u16* wt, const float* bqg, u16* qg) {
  int wave = blockIdx.x * 8 + (threadIdx.x >> 5);   // 48 waves
  int lane = threadIdx.x & 31;
  int col = lane & 15, hi = lane >> 4;
  int b_  = wave / 24;
  int rem = wave % 24;
  int mt = rem / 6, ng = rem % 6;
  int g = mt * 16 + col;
  int gp = gpos_of(g > GG - 1 ? GG - 1 : g);
  const u16* A  = tok + ((size_t)b_ * NN + gp) * TD;
  const u16* Wm = wt + (size_t)3 * TD * TD;

  v8f acc[4];
#pragma unroll
  for (int t = 0; t < 4; ++t) acc[t] = zero8();
  for (int kk = 0; kk < TD; kk += 32) {
    v16bf a = load_a16(A + kk, hi);
#pragma unroll
    for (int t = 0; t < 4; ++t) {
      v16bf b = load_b16(Wm + (size_t)(ng * 64 + t * 16 + col) * TD + kk, hi);
      acc[t] = wmma_bf(a, b, acc[t]);
    }
  }
#pragma unroll
  for (int t = 0; t < 4; ++t) {
#pragma unroll
    for (int r = 0; r < 8; ++r) {
      int ncol = ng * 64 + t * 16 + col;
      int m = mt * 16 + r + hi * 8;
      int d = ncol & 63;
      qg[(((size_t)b_ * HH + ng) * GP + m) * DD + d] = f2bf((acc[t][r] + bqg[ncol]) * 0.125f);
    }
  }
}

// ---------------- gather kg (B,H,GP,D) and vgT (B,H,D,GP), zero-padded ----------------
__global__ void gather_globals(const u16* k, const u16* vT, u16* kgb, u16* vgTb) {
  int idx = blockIdx.x * blockDim.x + threadIdx.x;  // B*H*GP*DD
  int d = idx & 63;
  int rest = idx >> 6;
  int g = rest & 63;
  int bh = rest >> 6;
  u16 kv = 0, vv = 0;
  if (g < GG) {
    int gp = gpos_of(g);
    kv = k[((size_t)bh * NN + gp) * DD + d];
    vv = vT[((size_t)bh * DD + d) * NN + gp];
  }
  kgb[((size_t)bh * GP + g) * DD + d]  = kv;
  vgTb[((size_t)bh * DD + d) * GP + g] = vv;
}

__global__ void map_init(int* map) {
  int i = blockIdx.x * blockDim.x + threadIdx.x;
  if (i < NN) map[i] = -1;
}
__global__ void map_set(int* map) {
  int g = threadIdx.x;
  if (g < GG) map[gpos_of(g)] = g;
}

// ---------------- banded + global-column attention, flash streaming ----------------
// Shared K/V tiles for the whole block are staged into LDS via async copies,
// double-buffered so the copy for tile kt+1 overlaps compute on tile kt.
__global__ void band_attn(const u16* q, const u16* k, const u16* vT,
                          const u16* kgb, const u16* vgTb, float* out_h) {
  __shared__ __align__(64) u16 pst[8][16][32];
  __shared__ __align__(64) u16 kbuf[2][32][64];   // [buf][key][d]
  __shared__ __align__(64) u16 vbuf[2][64][32];   // [buf][d][key]
  int lane = threadIdx.x & 31;
  int w = threadIdx.x >> 5;
  int col = lane & 15, hi = lane >> 4;
  int chunk = blockIdx.x % NC;
  int bh = blockIdx.x / NC;
  int q0 = chunk * W1 + w * 16;

  const u16* qrow = q + ((size_t)bh * NN + q0 + col) * DD;
  v16bf aq0 = load_a16(qrow, hi);
  v16bf aq1 = load_a16(qrow + 32, hi);

  v8f acc[4];
#pragma unroll
  for (int t = 0; t < 4; ++t) acc[t] = zero8();
  float rmax[8], rsum[8];
#pragma unroll
  for (int r = 0; r < 8; ++r) { rmax[r] = -3.0e38f; rsum[r] = 0.f; }

  for (int kt = 0; kt < 14; ++kt) {        // kt 0..1 globals (64 padded), 2..13 band (384)
    bool isg = kt < 2;
    // wait for async tile issued last iteration, then make buffer visible to all waves
    if (kt >= 2) wait_async0();
    __syncthreads();
    // issue async copy of next band tile into buf[(kt+1)&1]
    if (kt >= 1 && kt + 1 < 14) {
      int ktb = kt + 1;
      int bi = ktb & 1;
      int kb = chunk * W1 - W1 + (ktb - 2) * 32;
#pragma unroll
      for (int i = 0; i < 2; ++i) {
        int c = threadIdx.x * 2 + i;     // 512 x 16B chunks (8 KB)
        if (c < 256) {                    // K block: 32 rows x 128 B
          int row = c >> 3, part = c & 7;
          int ka = kb + row;
          ka = ka < 0 ? 0 : (ka > NN - 1 ? NN - 1 : ka);
          async_copy16(k + ((size_t)bh * NN + ka) * DD + part * 8,
                       &kbuf[bi][row][part * 8]);
        } else {                          // V block: 64 rows x 64 B (from vT)
          int idx = c - 256;
          int d = idx >> 2, part = idx & 3;
          int kbc = kb < 0 ? 0 : (kb > NN - 32 ? NN - 32 : kb);
          async_copy16(vT + ((size_t)bh * DD + d) * NN + kbc + part * 8,
                       &vbuf[bi][d][part * 8]);
        }
      }
    }
    // ----- scores S (16 x 32) -----
    v8f s[2];
#pragma unroll
    for (int st = 0; st < 2; ++st) {
      const u16* krow;
      if (isg) {
        int gidx = kt * 32 + st * 16 + col;
        krow = kgb + ((size_t)bh * GP + gidx) * DD;
      } else {
        krow = &kbuf[kt & 1][st * 16 + col][0];
      }
      v8f z = zero8();
      z = wmma_bf(aq0, load_b16(krow, hi), z);
      z = wmma_bf(aq1, load_b16(krow + 32, hi), z);
      s[st] = z;
    }
    // masks
#pragma unroll
    for (int st = 0; st < 2; ++st) {
#pragma unroll
      for (int r = 0; r < 8; ++r) {
        int m = r + hi * 8;
        int qi = w * 16 + m;
        bool valid;
        if (isg) {
          valid = (kt * 32 + st * 16 + col) < GG;
        } else {
          int jj = (kt - 2) * 32 + st * 16 + col;
          int ka = chunk * W1 - W1 + jj;
          int rel = jj - W1 - qi;
          valid = (rel >= -W1) && (rel <= W1) && (ka >= 0) && (ka < NN);
        }
        if (!valid) s[st][r] = NEGV;
      }
    }
    // row max of this 32-key tile
    float tm[8];
#pragma unroll
    for (int r = 0; r < 8; ++r) tm[r] = fmaxf(s[0][r], s[1][r]);
#pragma unroll
    for (int off = 1; off < 16; off <<= 1)
#pragma unroll
      for (int r = 0; r < 8; ++r) tm[r] = fmaxf(tm[r], __shfl_xor(tm[r], off, 32));
    // flash update
    float p0[8], p1[8], rp[8];
#pragma unroll
    for (int r = 0; r < 8; ++r) {
      float nm = fmaxf(rmax[r], tm[r]);
      float sc = __expf(rmax[r] - nm);
      rmax[r] = nm;
      p0[r] = __expf(s[0][r] - nm);
      p1[r] = __expf(s[1][r] - nm);
      rp[r] = p0[r] + p1[r];
      rsum[r] *= sc;
      acc[0][r] *= sc; acc[1][r] *= sc; acc[2][r] *= sc; acc[3][r] *= sc;
    }
#pragma unroll
    for (int off = 1; off < 16; off <<= 1)
#pragma unroll
      for (int r = 0; r < 8; ++r) rp[r] += __shfl_xor(rp[r], off, 32);
#pragma unroll
    for (int r = 0; r < 8; ++r) rsum[r] += rp[r];
    // stage P (C-layout -> A-fragment) through LDS
    __syncthreads();
#pragma unroll
    for (int r = 0; r < 8; ++r) {
      pst[w][r + hi * 8][col]      = f2bf(p0[r]);
      pst[w][r + hi * 8][16 + col] = f2bf(p1[r]);
    }
    __syncthreads();
    v16bf ap = load_a16(&pst[w][col][0], hi);
    // P @ V
#pragma unroll
    for (int t = 0; t < 4; ++t) {
      int d = t * 16 + col;
      const u16* vrun;
      if (isg) vrun = vgTb + ((size_t)bh * DD + d) * GP + kt * 32;
      else     vrun = &vbuf[kt & 1][d][0];
      acc[t] = wmma_bf(ap, load_b16(vrun, hi), acc[t]);
    }
  }
  // epilogue
#pragma unroll
  for (int t = 0; t < 4; ++t)
#pragma unroll
    for (int r = 0; r < 8; ++r)
      out_h[((size_t)bh * NN + q0 + r + hi * 8) * DD + t * 16 + col] = acc[t][r] / rsum[r];
}

// ---------------- global attention: qg vs all N keys ----------------
__global__ void global_attn(const u16* qg, const u16* kga, const u16* vgaT, float* og) {
  __shared__ __align__(64) u16 pst[8][16][32];
  __shared__ float lacc[8][16][64];
  __shared__ float lmv[8][16];
  __shared__ float llv[8][16];
  int lane = threadIdx.x & 31;
  int w = threadIdx.x >> 5;
  int col = lane & 15, hi = lane >> 4;
  int mt = blockIdx.x & 3;
  int bh = blockIdx.x >> 2;

  const u16* qrow = qg + ((size_t)bh * GP + mt * 16 + col) * DD;
  v16bf aq0 = load_a16(qrow, hi);
  v16bf aq1 = load_a16(qrow + 32, hi);

  v8f acc[4];
#pragma unroll
  for (int t = 0; t < 4; ++t) acc[t] = zero8();
  float rmax[8], rsum[8];
#pragma unroll
  for (int r = 0; r < 8; ++r) { rmax[r] = -3.0e38f; rsum[r] = 0.f; }

  int key_base = w * (NN / 8);
  for (int it = 0; it < (NN / 8) / 32; ++it) {
    int k0 = key_base + it * 32;
    v8f s[2];
#pragma unroll
    for (int st = 0; st < 2; ++st) {
      const u16* krow = kga + ((size_t)bh * NN + k0 + st * 16 + col) * DD;
      v8f z = zero8();
      z = wmma_bf(aq0, load_b16(krow, hi), z);
      z = wmma_bf(aq1, load_b16(krow + 32, hi), z);
      s[st] = z;
    }
    float tm[8];
#pragma unroll
    for (int r = 0; r < 8; ++r) tm[r] = fmaxf(s[0][r], s[1][r]);
#pragma unroll
    for (int off = 1; off < 16; off <<= 1)
#pragma unroll
      for (int r = 0; r < 8; ++r) tm[r] = fmaxf(tm[r], __shfl_xor(tm[r], off, 32));
    float p0[8], p1[8], rp[8];
#pragma unroll
    for (int r = 0; r < 8; ++r) {
      float nm = fmaxf(rmax[r], tm[r]);
      float sc = __expf(rmax[r] - nm);
      rmax[r] = nm;
      p0[r] = __expf(s[0][r] - nm);
      p1[r] = __expf(s[1][r] - nm);
      rp[r] = p0[r] + p1[r];
      rsum[r] *= sc;
      acc[0][r] *= sc; acc[1][r] *= sc; acc[2][r] *= sc; acc[3][r] *= sc;
    }
#pragma unroll
    for (int off = 1; off < 16; off <<= 1)
#pragma unroll
      for (int r = 0; r < 8; ++r) rp[r] += __shfl_xor(rp[r], off, 32);
#pragma unroll
    for (int r = 0; r < 8; ++r) rsum[r] += rp[r];
    __syncthreads();
#pragma unroll
    for (int r = 0; r < 8; ++r) {
      pst[w][r + hi * 8][col]      = f2bf(p0[r]);
      pst[w][r + hi * 8][16 + col] = f2bf(p1[r]);
    }
    __syncthreads();
    v16bf ap = load_a16(&pst[w][col][0], hi);
#pragma unroll
    for (int t = 0; t < 4; ++t) {
      const u16* vrun = vgaT + ((size_t)bh * DD + t * 16 + col) * NN + k0;
      acc[t] = wmma_bf(ap, load_b16(vrun, hi), acc[t]);
    }
  }
  // write wave partials
#pragma unroll
  for (int t = 0; t < 4; ++t)
#pragma unroll
    for (int r = 0; r < 8; ++r)
      lacc[w][r + hi * 8][t * 16 + col] = acc[t][r];
  if (col == 0) {
#pragma unroll
    for (int r = 0; r < 8; ++r) { lmv[w][r + hi * 8] = rmax[r]; llv[w][r + hi * 8] = rsum[r]; }
  }
  __syncthreads();
  // merge 8 wave partials
  for (int e = threadIdx.x; e < 16 * 64; e += 256) {
    int m = e >> 6, d = e & 63;
    float mt_ = -3.0e38f;
#pragma unroll
    for (int w2 = 0; w2 < 8; ++w2) mt_ = fmaxf(mt_, lmv[w2][m]);
    float den = 0.f, num = 0.f;
#pragma unroll
    for (int w2 = 0; w2 < 8; ++w2) {
      float f = __expf(lmv[w2][m] - mt_);
      den += llv[w2][m] * f;
      num += lacc[w2][m][d] * f;
    }
    og[((size_t)bh * GP + mt * 16 + m) * DD + d] = num / den;
  }
}

// ---------------- untokenize (with GPOS override) ----------------
__global__ void untokenize(const float* out_h, const float* og, const int* map, float* out) {
  int idx = blockIdx.x * blockDim.x + threadIdx.x;   // B*C*IMG*IMG
  int xx = idx % IMGS;
  int rest = idx / IMGS;
  int yy = rest % IMGS;
  rest /= IMGS;
  int c = rest % CC;
  int b = rest / CC;
  int i = yy >> 1, wi = yy & 1, j = xx >> 1, wj = xx & 1;
  int n = i * NS + j;
  int td = c * 4 + wi * 2 + wj;
  int h = td >> 6, d = td & 63;
  int g = map[n];
  size_t bh = (size_t)b * HH + h;
  float v = (g >= 0) ? og[(bh * GP + g) * DD + d] : out_h[(bh * NN + n) * DD + d];
  out[idx] = v;
}

extern "C" void kernel_launch(void* const* d_in, const int* in_sizes, int n_in,
                              void* d_out, int out_size, void* d_ws, size_t ws_size,
                              hipStream_t stream) {
  (void)in_sizes; (void)n_in; (void)out_size; (void)ws_size;
  const float* x   = (const float*)d_in[0];
  const float* Wq  = (const float*)d_in[1];
  const float* bq  = (const float*)d_in[2];
  const float* Wk  = (const float*)d_in[3];
  const float* bk  = (const float*)d_in[4];
  const float* Wv  = (const float*)d_in[5];
  const float* bv  = (const float*)d_in[6];
  const float* Wqg = (const float*)d_in[7];
  const float* bqg = (const float*)d_in[8];
  const float* Wkg = (const float*)d_in[9];
  const float* bkg = (const float*)d_in[10];
  const float* Wvg = (const float*)d_in[11];
  const float* bvg = (const float*)d_in[12];

  char* ws = (char*)d_ws;
  size_t off = 0;
  auto alloc = [&](size_t bytes) -> void* {
    void* p = ws + off;
    off = (off + bytes + 255) & ~(size_t)255;
    return p;
  };
  u16* tok   = (u16*)alloc((size_t)2 * BB * NN * TD);
  u16* wt    = (u16*)alloc((size_t)2 * 6 * TD * TD);
  u16* qb    = (u16*)alloc((size_t)2 * BB * HH * NN * DD);
  u16* kb    = (u16*)alloc((size_t)2 * BB * HH * NN * DD);
  u16* vTb   = (u16*)alloc((size_t)2 * BB * HH * NN * DD);
  u16* kgab  = (u16*)alloc((size_t)2 * BB * HH * NN * DD);
  u16* vgaTb = (u16*)alloc((size_t)2 * BB * HH * NN * DD);
  u16* kgbb  = (u16*)alloc((size_t)2 * BB * HH * GP * DD);
  u16* vgTbb = (u16*)alloc((size_t)2 * BB * HH * GP * DD);
  u16* qgb   = (u16*)alloc((size_t)2 * BB * HH * GP * DD);
  float* outh = (float*)alloc((size_t)4 * BB * HH * NN * DD);
  float* ogb  = (float*)alloc((size_t)4 * BB * HH * GP * DD);
  int* map    = (int*)alloc((size_t)4 * NN);

  prep_weights<<<(6 * TD * TD) / 256, 256, 0, stream>>>(Wq, Wk, Wv, Wqg, Wkg, Wvg, wt);
  tokenize<<<(BB * NN * TD) / 256, 256, 0, stream>>>(x, tok);
  proj_gemm<<<7680, 256, 0, stream>>>(tok, wt, bq, bk, bv, bkg, bvg,
                                      qb, kb, vTb, kgab, vgaTb);
  qg_gemm<<<6, 256, 0, stream>>>(tok, wt, bqg, qgb);
  gather_globals<<<(BB * HH * GP * DD) / 256, 256, 0, stream>>>(kb, vTb, kgbb, vgTbb);
  map_init<<<NN / 256, 256, 0, stream>>>(map);
  map_set<<<1, 64, 0, stream>>>(map);
  band_attn<<<BB * HH * NC, 256, 0, stream>>>(qb, kb, vTb, kgbb, vgTbb, outh);
  global_attn<<<BB * HH * 4, 256, 0, stream>>>(qgb, kgab, vgaTb, ogb);
  untokenize<<<(BB * CC * IMGS * IMGS) / 256, 256, 0, stream>>>(outh, ogb, map, (float*)d_out);
}
